// REWAVisionAttention_46557445489438
// MI455X (gfx1250) — compile-verified, matrix operand order checked by reading, not run on hardware
//
#include <hip/hip_runtime.h>
#include <hip/hip_bf16.h>

typedef _Float16 v16h __attribute__((ext_vector_type(16)));
typedef float    v8f  __attribute__((ext_vector_type(8)));
typedef _Float16 h2v  __attribute__((ext_vector_type(2)));

namespace cfg {
constexpr int B = 4, N = 1024, C = 512, H = 8, DH = 64, INNER = 512;
constexpr int DE = 64, DS = 64, DHY = 32;
constexpr int ROWS = B * H * N;   // 32768 (b,h,n) rows
constexpr int TOK  = B * N;       // 4096 tokens
}

// K index of element-pair v (v=0..7) within a 16-bit WMMA A/B operand (ISA 7.12.2)
__device__ __forceinline__ int kmap_pair(int v, int half) {
  return ((v & 4) ? 16 : 0) + (v & 3) * 2 + half * 8;
}

// Load one 16(row)x32(K) f16 operand tile; `row` is the lane's row (A) or column (B),
// matrix is row-major with leading dim `ld` (elements). The pair pattern merges into
// two global_load_b128 per tile (verified in round-1 disassembly).
__device__ __forceinline__ v16h load_tile_f16(const _Float16* __restrict__ p, int ld,
                                              int row, int k0, int half) {
  v16h a;
  const _Float16* base = p + (long)row * ld + k0;
#pragma unroll
  for (int v = 0; v < 8; ++v) {
    int K = kmap_pair(v, half);
    h2v pr = *(const h2v*)(base + K);
    a[2 * v]     = pr[0];
    a[2 * v + 1] = pr[1];
  }
  return a;
}

__device__ __forceinline__ v8f wmma_f16(v16h a, v16h b, v8f c) {
  return __builtin_amdgcn_wmma_f32_16x16x32_f16(false, a, false, b, (short)0, c,
                                                false, false);
}

__device__ __forceinline__ float softplusf(float x) { return log1pf(__expf(x)); }

// flat shared-pointer -> wave-relative LDS byte offset (ISA 10.2: LDS aperture
// flat address bits [31:0] are the LDS offset)
__device__ __forceinline__ unsigned lds_offset(const void* p) {
  return (unsigned)(unsigned long long)p;
}

// async wave-wide copy of one 16B segment per lane: global -> LDS (ASYNCcnt path)
__device__ __forceinline__ void async_copy_b128(unsigned lds_addr, const void* gaddr) {
  asm volatile("global_load_async_to_lds_b128 %0, %1, off"
               :: "v"(lds_addr), "v"(gaddr) : "memory");
}
__device__ __forceinline__ void wait_asynccnt0() {
  asm volatile("s_wait_asynccnt 0x0" ::: "memory");
}

// ---------------------------------------------------------------- utilities
__global__ void k_f32_to_f16(const float* __restrict__ s, _Float16* __restrict__ d, int n) {
  int i = blockIdx.x * blockDim.x + threadIdx.x;
  if (i < n) d[i] = (_Float16)s[i];
}

__global__ void k_zero_f32(float* __restrict__ p, int n) {
  int i = blockIdx.x * blockDim.x + threadIdx.x;
  if (i < n) p[i] = 0.0f;
}

// qkv (TOK x 3*INNER) f16 -> q/k/v in (B,H,N,DH) f16
__global__ void k_repack_qkv(const _Float16* __restrict__ qkvh,
                             _Float16* __restrict__ qh, _Float16* __restrict__ kh,
                             _Float16* __restrict__ vh) {
  using namespace cfg;
  int idx = blockIdx.x * blockDim.x + threadIdx.x;  // over ROWS*DH
  if (idx >= ROWS * DH) return;
  int d = idx & 63;
  int rem = idx >> 6;           // (b*H+h)*N + n
  int n = rem & (N - 1);
  int bh = rem >> 10;
  int h = bh & (H - 1), b = bh >> 3;
  long src = (long)(b * N + n) * (3 * INNER) + h * DH + d;
  long dst = idx;
  qh[dst] = qkvh[src];
  kh[dst] = qkvh[src + INNER];
  vh[dst] = qkvh[src + 2 * INNER];
}

// normalize rows of f16 matrix by sqrt(sumsq[row]) (spherical embeddings)
__global__ void k_normalize(_Float16* __restrict__ s, const float* __restrict__ sumsq,
                            int rows, int cols) {
  int i = blockIdx.x * blockDim.x + threadIdx.x;
  if (i >= rows * cols) return;
  int row = i / cols;
  float inv = 1.0f / fmaxf(sqrtf(sumsq[row]), 1e-12f);
  s[i] = (_Float16)((float)s[i] * inv);
}

// --------------------------------------------------------- generic WMMA GEMM
// out = A(MxK) * W(NoutxK)^T + bias ; one wave per 16x16 C tile.
// MODE 0: f32 out            (projection, writes d_out)
// MODE 1: f16 out            (qkv)
// MODE 2: f16 out + atomic row-sumsq        (euclid / sphere embeds)
// MODE 3: tanh -> f16 out + atomic row-sumsq (hyperbolic embeds)
template <int MODE>
__global__ void wmma_gemm(const _Float16* __restrict__ A, const _Float16* __restrict__ W,
                          const float* __restrict__ bias, void* __restrict__ out,
                          float* __restrict__ sumsq, int M, int Nout, int K) {
  int wave = (blockIdx.x * blockDim.x + threadIdx.x) >> 5;
  int lane = threadIdx.x & 31;
  int ntn = Nout >> 4;
  int tm = wave / ntn, tn = wave % ntn;
  if (tm >= (M >> 4)) return;  // wave-uniform
  int half = lane >> 4, l16 = lane & 15;
  int arow = tm * 16 + l16;    // A row this lane feeds
  int wrow = tn * 16 + l16;    // W row == output column this lane feeds
  v8f c;
  float bv = bias ? bias[wrow] : 0.0f;  // bias is per-column -> per-lane, all 8 rows
#pragma unroll
  for (int r = 0; r < 8; ++r) c[r] = bv;
  for (int k0 = 0; k0 < K; k0 += 32) {
    if (k0 + 32 < K) {
      __builtin_prefetch(A + (long)arow * K + k0 + 32, 0, 3);
      __builtin_prefetch(W + (long)wrow * K + k0 + 32, 0, 3);
    }
    v16h a = load_tile_f16(A, K, arow, k0, half);
    v16h b = load_tile_f16(W, K, wrow, k0, half);
    c = wmma_f16(a, b, c);
  }
#pragma unroll
  for (int r = 0; r < 8; ++r) {
    int row = tm * 16 + r + 8 * half;
    int col = tn * 16 + l16;
    float y = c[r];
    if (MODE == 3) y = tanhf(y);
    if (MODE == 0)
      ((float*)out)[(long)row * Nout + col] = y;
    else
      ((_Float16*)out)[(long)row * Nout + col] = (_Float16)y;
    if (MODE >= 2) {
      float ss = y * y;  // reduce across the 16 lanes holding this row
#pragma unroll
      for (int off = 1; off < 16; off <<= 1) ss += __shfl_xor(ss, off, 32);
      if (l16 == 0) atomicAdd(&sumsq[row], ss);
    }
  }
}

// -------------------------------------------------- fused multi-metric attention
// One wave per 16-row query tile; online softmax over 32-column key blocks.
// V tiles are staged global->LDS with wave-wide async b128 copies (ASYNCcnt),
// issued before the score WMMAs so the DMA overlaps the distance/softmax VALU work.
__global__ void __launch_bounds__(32) k_attn(
    const _Float16* __restrict__ eqh, const _Float16* __restrict__ ekh,
    const _Float16* __restrict__ sqh, const _Float16* __restrict__ skh,
    const _Float16* __restrict__ hqh, const _Float16* __restrict__ hkh,
    const _Float16* __restrict__ vmat,
    const float* __restrict__ eqn, const float* __restrict__ ekn,
    const float* __restrict__ un, const float* __restrict__ vnrm,
    const float* __restrict__ pAl, const float* __restrict__ pBe,
    const float* __restrict__ pGa, const float* __restrict__ pTe,
    _Float16* __restrict__ o16) {
  using namespace cfg;
  __shared__ _Float16 ldsP[16 * 32];   // P tile staging: D-layout -> A-layout transpose
  __shared__ _Float16 ldsV[32 * 64];   // async-staged V tile (4 KB, contiguous in HBM)
  const int lane = threadIdx.x & 31;
  const int half = lane >> 4, l16 = lane & 15;
  const int tile = blockIdx.x;
  const int itile = tile & (N / 16 - 1);
  const int bh = tile >> 6;  // N/16 == 64
  const long eb = (long)bh * N * 64;
  const long hb = (long)bh * N * 32;
  const long nb0 = (long)bh * N;
  const unsigned ldsVbase = lds_offset(ldsV);

  const float ca = softplusf(pAl[0]);
  const float cb = softplusf(pBe[0]);
  const float cg = softplusf(pGa[0]);
  const float invT = 1.0f / softplusf(pTe[0]);

  const int arow = itile * 16 + l16;
  const v16h aE0 = load_tile_f16(eqh + eb, 64, arow, 0, half);
  const v16h aE1 = load_tile_f16(eqh + eb, 64, arow, 32, half);
  const v16h aS0 = load_tile_f16(sqh + eb, 64, arow, 0, half);
  const v16h aS1 = load_tile_f16(sqh + eb, 64, arow, 32, half);
  const v16h aH  = load_tile_f16(hqh + hb, 32, arow, 0, half);

  float rEn[8], rUn[8], mrow[8], lrow[8];
  v8f o0 = {}, o1 = {}, o2 = {}, o3 = {};
#pragma unroll
  for (int r = 0; r < 8; ++r) {
    int row = itile * 16 + r + 8 * half;  // rows this lane holds in D layout
    rEn[r] = eqn[nb0 + row];
    rUn[r] = un[nb0 + row];
    mrow[r] = -1e30f;
    lrow[r] = 0.0f;
  }

  for (int jb = 0; jb < N / 32; ++jb) {
    // ---- kick off async DMA of the 32x64 V tile (contiguous 4 KB) into LDS
    {
      const char* gtile = (const char*)(vmat + eb + (long)jb * 32 * 64);
#pragma unroll
      for (int i2 = 0; i2 < 8; ++i2) {
        int seg = i2 * 32 + lane;  // 256 x 16B segments
        async_copy_b128(ldsVbase + seg * 16, gtile + (size_t)seg * 16);
      }
    }
    // prefetch next block's key tiles while we compute
    if (jb + 1 < N / 32) {
      __builtin_prefetch(ekh + eb + (long)((jb + 1) * 32 + lane) * 64, 0, 3);
      __builtin_prefetch(skh + eb + (long)((jb + 1) * 32 + lane) * 64, 0, 3);
    }
    float sc[2][8];
#pragma unroll
    for (int jt = 0; jt < 2; ++jt) {
      const int wrow = jb * 32 + jt * 16 + l16;  // key index == output column
      v16h bE0 = load_tile_f16(ekh + eb, 64, wrow, 0, half);
      v16h bE1 = load_tile_f16(ekh + eb, 64, wrow, 32, half);
      v16h bS0 = load_tile_f16(skh + eb, 64, wrow, 0, half);
      v16h bS1 = load_tile_f16(skh + eb, 64, wrow, 32, half);
      v16h bH  = load_tile_f16(hkh + hb, 32, wrow, 0, half);
      v8f se = {}, ssim = {}, shy = {};
      se = wmma_f16(aE0, bE0, se);
      se = wmma_f16(aE1, bE1, se);
      ssim = wmma_f16(aS0, bS0, ssim);
      ssim = wmma_f16(aS1, bS1, ssim);
      shy = wmma_f16(aH, bH, shy);
      const float cEn = ekn[nb0 + wrow];   // column stats: per-lane scalars
      const float cVn = vnrm[nb0 + wrow];
#pragma unroll
      for (int r = 0; r < 8; ++r) {
        float dE = sqrtf(fmaxf(rEn[r] + cEn - 2.0f * se[r], 1e-12f));
        float sim = fminf(fmaxf(ssim[r], -1.0f + 1e-6f), 1.0f - 1e-6f);
        float dSs = acosf(sim);
        float dns = fmaxf(rUn[r] + cVn - 2.0f * shy[r], 0.0f);
        float den = (1.0f - rUn[r]) * (1.0f - cVn);
        float arg = fmaxf(1.0f + 2.0f * dns / (den + 1e-8f), 1.0f + 1e-6f);
        float dHh = acoshf(arg);
        sc[jt][r] = -(ca * dE + cb * dSs + cg * dHh) * invT;
      }
    }
    // ---- online softmax across this 32-column block (row = 16-lane group)
    float p0[8], p1[8];
#pragma unroll
    for (int r = 0; r < 8; ++r) {
      float tmax = fmaxf(sc[0][r], sc[1][r]);
#pragma unroll
      for (int off = 1; off < 16; off <<= 1)
        tmax = fmaxf(tmax, __shfl_xor(tmax, off, 32));
      float mnew = fmaxf(mrow[r], tmax);
      float corr = __expf(mrow[r] - mnew);
      float e0 = __expf(sc[0][r] - mnew);
      float e1 = __expf(sc[1][r] - mnew);
      float rs = e0 + e1;
#pragma unroll
      for (int off = 1; off < 16; off <<= 1) rs += __shfl_xor(rs, off, 32);
      lrow[r] = lrow[r] * corr + rs;
      mrow[r] = mnew;
      p0[r] = e0;
      p1[r] = e1;
      o0[r] *= corr; o1[r] *= corr; o2[r] *= corr; o3[r] *= corr;
    }
    // ---- stage P through LDS: D-layout (col=lane) -> A-layout (row=lane)
#pragma unroll
    for (int r = 0; r < 8; ++r) {
      int rowm = r + 8 * half;
      ldsP[rowm * 32 + l16]      = (_Float16)p0[r];
      ldsP[rowm * 32 + 16 + l16] = (_Float16)p1[r];
    }
    __syncthreads();
    v16h pa;
#pragma unroll
    for (int v = 0; v < 8; ++v) {
      int K = kmap_pair(v, half);
      h2v pr = *(const h2v*)&ldsP[l16 * 32 + K];
      pa[2 * v] = pr[0];
      pa[2 * v + 1] = pr[1];
    }
    __syncthreads();
    // ---- O(16x64) += P(16x32) * V(32x64); V comes from the async-staged LDS tile
    wait_asynccnt0();
#pragma unroll
    for (int nb = 0; nb < 4; ++nb) {
      v16h bv;
#pragma unroll
      for (int v = 0; v < 8; ++v) {
        int K = kmap_pair(v, half);
        bv[2 * v]     = ldsV[K * 64 + nb * 16 + l16];
        bv[2 * v + 1] = ldsV[(K + 1) * 64 + nb * 16 + l16];
      }
      v8f acc = (nb == 0) ? o0 : (nb == 1) ? o1 : (nb == 2) ? o2 : o3;
      acc = wmma_f16(pa, bv, acc);
      if (nb == 0) o0 = acc; else if (nb == 1) o1 = acc;
      else if (nb == 2) o2 = acc; else o3 = acc;
    }
  }
  // ---- finalize, write f16 in (B, N, H*DH) token-major layout for projection
  const int b = bh >> 3, h = bh & (H - 1);
#pragma unroll
  for (int r = 0; r < 8; ++r) {
    const float inv = 1.0f / lrow[r];
    const int row = itile * 16 + r + 8 * half;
    const long ob = (((long)(b * N + row)) * H + h) * 64 + l16;
    o16[ob + 0]  = (_Float16)(o0[r] * inv);
    o16[ob + 16] = (_Float16)(o1[r] * inv);
    o16[ob + 32] = (_Float16)(o2[r] * inv);
    o16[ob + 48] = (_Float16)(o3[r] * inv);
  }
}

// ------------------------------------------------------------------ launcher
extern "C" void kernel_launch(void* const* d_in, const int* in_sizes, int n_in,
                              void* d_out, int out_size, void* d_ws, size_t ws_size,
                              hipStream_t stream) {
  using namespace cfg;
  (void)in_sizes; (void)n_in; (void)out_size; (void)ws_size;
  const float* x      = (const float*)d_in[0];
  const float* w_qkv  = (const float*)d_in[1];
  const float* w_proj = (const float*)d_in[2];
  const float* b_proj = (const float*)d_in[3];
  const float* w_e    = (const float*)d_in[4];
  const float* b_e    = (const float*)d_in[5];
  const float* w_s    = (const float*)d_in[6];
  const float* b_s    = (const float*)d_in[7];
  const float* w_h    = (const float*)d_in[8];
  const float* b_h    = (const float*)d_in[9];
  const float* alpha  = (const float*)d_in[10];
  const float* beta   = (const float*)d_in[11];
  const float* gamma  = (const float*)d_in[12];
  const float* temp   = (const float*)d_in[13];

  char* p = (char*)d_ws;
  auto take = [&p](size_t bytes) -> char* {
    char* r = p;
    p += (bytes + 255) & ~(size_t)255;
    return r;
  };
  _Float16* xh     = (_Float16*)take((size_t)TOK * C * 2);
  _Float16* wqkvh  = (_Float16*)take((size_t)3 * INNER * C * 2);
  _Float16* wprojh = (_Float16*)take((size_t)C * INNER * 2);
  _Float16* weh    = (_Float16*)take((size_t)DE * DH * 2);
  _Float16* wsh    = (_Float16*)take((size_t)DS * DH * 2);
  _Float16* whh    = (_Float16*)take((size_t)DHY * DH * 2);
  _Float16* qkvh   = (_Float16*)take((size_t)TOK * 3 * INNER * 2);
  _Float16* qh     = (_Float16*)take((size_t)ROWS * DH * 2);
  _Float16* kh     = (_Float16*)take((size_t)ROWS * DH * 2);
  _Float16* vh     = (_Float16*)take((size_t)ROWS * DH * 2);
  _Float16* eqh    = (_Float16*)take((size_t)ROWS * DE * 2);
  _Float16* ekh    = (_Float16*)take((size_t)ROWS * DE * 2);
  _Float16* sqh    = (_Float16*)take((size_t)ROWS * DS * 2);
  _Float16* skh    = (_Float16*)take((size_t)ROWS * DS * 2);
  _Float16* hqh    = (_Float16*)take((size_t)ROWS * DHY * 2);
  _Float16* hkh    = (_Float16*)take((size_t)ROWS * DHY * 2);
  float*    norms  = (float*)take((size_t)6 * ROWS * 4);
  _Float16* o16    = (_Float16*)take((size_t)TOK * INNER * 2);
  float* eqn = norms;
  float* ekn = norms + ROWS;
  float* sqn = norms + 2 * ROWS;
  float* skn = norms + 3 * ROWS;
  float* unb = norms + 4 * ROWS;
  float* vnb = norms + 5 * ROWS;

  auto conv = [&](const float* s, _Float16* d, int n) {
    k_f32_to_f16<<<(n + 255) / 256, 256, 0, stream>>>(s, d, n);
  };
  conv(x, xh, TOK * C);
  conv(w_qkv, wqkvh, 3 * INNER * C);
  conv(w_proj, wprojh, C * INNER);
  conv(w_e, weh, DE * DH);
  conv(w_s, wsh, DS * DH);
  conv(w_h, whh, DHY * DH);

  k_zero_f32<<<(6 * ROWS + 255) / 256, 256, 0, stream>>>(norms, 6 * ROWS);

  {  // QKV GEMM: (TOK x C) @ (3*INNER x C)^T
    int waves = (TOK / 16) * ((3 * INNER) / 16);
    wmma_gemm<1><<<waves * 32 / 256, 256, 0, stream>>>(xh, wqkvh, nullptr, qkvh,
                                                       nullptr, TOK, 3 * INNER, C);
  }
  k_repack_qkv<<<(ROWS * DH + 255) / 256, 256, 0, stream>>>(qkvh, qh, kh, vh);

  {  // metric embedding GEMMs with fused bias / tanh / row-sumsq epilogues
    int wavesE = (ROWS / 16) * (DE / 16);
    wmma_gemm<2><<<wavesE * 32 / 256, 256, 0, stream>>>(qh, weh, b_e, eqh, eqn, ROWS, DE, DH);
    wmma_gemm<2><<<wavesE * 32 / 256, 256, 0, stream>>>(kh, weh, b_e, ekh, ekn, ROWS, DE, DH);
    wmma_gemm<2><<<wavesE * 32 / 256, 256, 0, stream>>>(qh, wsh, b_s, sqh, sqn, ROWS, DS, DH);
    wmma_gemm<2><<<wavesE * 32 / 256, 256, 0, stream>>>(kh, wsh, b_s, skh, skn, ROWS, DS, DH);
    int wavesH = (ROWS / 16) * (DHY / 16);
    wmma_gemm<3><<<wavesH * 32 / 256, 256, 0, stream>>>(qh, whh, b_h, hqh, unb, ROWS, DHY, DH);
    wmma_gemm<3><<<wavesH * 32 / 256, 256, 0, stream>>>(kh, whh, b_h, hkh, vnb, ROWS, DHY, DH);
  }
  k_normalize<<<(ROWS * DS + 255) / 256, 256, 0, stream>>>(sqh, sqn, ROWS, DS);
  k_normalize<<<(ROWS * DS + 255) / 256, 256, 0, stream>>>(skh, skn, ROWS, DS);

  k_attn<<<B * H * (N / 16), 32, 0, stream>>>(eqh, ekh, sqh, skh, hqh, hkh, vh,
                                              eqn, ekn, unb, vnb,
                                              alpha, beta, gamma, temp, o16);

  {  // output projection: (TOK x INNER) @ (C x INNER)^T + b_proj -> f32 d_out
    int waves = (TOK / 16) * (C / 16);
    wmma_gemm<0><<<waves * 32 / 256, 256, 0, stream>>>(o16, wprojh, b_proj,
                                                       d_out, nullptr, TOK, C, INNER);
  }
}
